// GAP_39814346834350
// MI455X (gfx1250) — matrix operand order repaired and checked by reading, not compiled, not run on hardware
//
#include <hip/hip_runtime.h>

typedef __bf16 bf16;
typedef __attribute__((ext_vector_type(16))) __bf16 v16bf;
typedef __attribute__((ext_vector_type(8)))  __bf16 v8bf;
typedef __attribute__((ext_vector_type(4)))  __bf16 v4bf;
typedef __attribute__((ext_vector_type(8)))  float  v8f;
typedef __attribute__((ext_vector_type(4)))  float  v4f;
typedef __attribute__((ext_vector_type(4)))  int    v4i;

#define NN    16384
#define FEAT  500
#define FPAD  512
#define K1    1024
#define EMB   128
#define BATCH 4096
#define NCLS  10
#define LSTR  40   // LDS row stride (bf16 elems): even, padded vs 32, 16B-aligned rows
#define KCH   2048 // split-K chunk for gemm3 (8 chunks)

#define AS1 __attribute__((address_space(1)))
#define AS3 __attribute__((address_space(3)))

#if __has_builtin(__builtin_amdgcn_global_load_async_to_lds_b128) && \
    __has_builtin(__builtin_amdgcn_s_wait_asynccnt)
#define HAVE_ASYNC_LDS 1
#else
#define HAVE_ASYNC_LDS 0
#endif

#if __has_builtin(__builtin_amdgcn_sched_barrier)
#define SCHED_FENCE() __builtin_amdgcn_sched_barrier(0)
#else
#define SCHED_FENCE()
#endif

// ---------------------------------------------------------------------------
// A fragment (16-bit A 16x32): lanes 0-15 hold K {0..7,16..23}; 16-31 +8.
// ---------------------------------------------------------------------------
__device__ __forceinline__ v16bf load_a_frag(const bf16* __restrict__ As,
                                             int wave, int lane) {
    const int half = lane >> 4;
    const int lm   = lane & 15;
    v16bf a;
    const bf16* ar = As + (wave * 16 + lm) * LSTR;
#pragma unroll
    for (int j = 0; j < 8; ++j) {
        const int kb = ((j < 4) ? (2 * j) : (16 + 2 * (j - 4))) + (half ? 8 : 0);
        a[2 * j]     = ar[kb];
        a[2 * j + 1] = ar[kb + 1];
    }
    return a;
}

// 8 n-tiles (128 cols) of WMMA against a prebuilt A fragment. All 16
// ds_load_b128 issue before the 8 back-to-back v_wmma (sched_barrier).
__device__ __forceinline__ void mma8(v16bf a, const bf16* __restrict__ Bsg,
                                     v8f* acc, int lane) {
    const int half = lane >> 4;
    const int lm   = lane & 15;
    v16bf b[8];
#pragma unroll
    for (int t = 0; t < 8; ++t) {
        const bf16* br = Bsg + (t * 16 + lm) * LSTR + half * 8;
#pragma unroll
        for (int j = 0; j < 16; ++j) b[t][j] = br[j + (j & 8)];
    }
    SCHED_FENCE();
#pragma unroll
    for (int t = 0; t < 8; ++t)
        acc[t] = __builtin_amdgcn_wmma_f32_16x16x32_bf16(
            false, a, false, b[t], (short)0, acc[t], false, false);
}

// Stage B tile (32 x NB cols) transposed into Bs[n][k]; wide b128 reads.
template <int NB>
__device__ __forceinline__ void stage_b_t(const bf16* __restrict__ B, int ldb,
                                          bf16* Bs, int kb, int nb, int tid) {
#pragma unroll
    for (int it = 0; it < (32 * NB) / (256 * 8); ++it) {
        const int li = (it * 256 + tid) * 8;
        const int k = li / NB, n = li % NB;
        const v8bf q = *(const v8bf*)(B + (size_t)(kb + k) * ldb + (nb + n));
#pragma unroll
        for (int j = 0; j < 8; ++j) Bs[(n + j) * LSTR + k] = q[j];
    }
}

// Stage A tile (128 x 32) bf16 -> bf16 (layout-preserving), manual fallback.
__device__ __forceinline__ void stage_a_bf(const bf16* __restrict__ A, int lda,
                                           bf16* As, int mb, int kb, int tid) {
#pragma unroll
    for (int it = 0; it < 2; ++it) {
        const int li = (it * 256 + tid) * 8;
        const int r = li >> 5, c = li & 31;
        *(v8bf*)(As + r * LSTR + c) = *(const v8bf*)(A + (size_t)(mb + r) * lda + (kb + c));
    }
}

// ---------------------------------------------------------------------------
__global__ __launch_bounds__(256) void deg_kernel(const float* __restrict__ adj,
                                                  float* __restrict__ deg) {
    __shared__ float red[256];
    const int row = blockIdx.x;
    float s = 0.0f;
    for (int c = threadIdx.x; c < NN; c += 256)
        s += adj[(size_t)row * NN + c];
    red[threadIdx.x] = s;
    __syncthreads();
    for (int off = 128; off > 0; off >>= 1) {
        if (threadIdx.x < off) red[threadIdx.x] += red[threadIdx.x + off];
        __syncthreads();
    }
    if (threadIdx.x == 0) deg[row] = red[0];
}

__global__ __launch_bounds__(256) void pack_data_kernel(const float* __restrict__ data,
                                                        bf16* __restrict__ X1) {
    const int li  = blockIdx.x * 256 + threadIdx.x;   // over NN*FPAD
    const int row = li >> 9;
    const int c   = li & 511;
    const float v = (c < FEAT) ? data[(size_t)row * FEAT + c] : 0.0f;
    X1[(size_t)row * K1 + c] = (bf16)v;
}

__global__ __launch_bounds__(256) void pack_w1_kernel(const float* __restrict__ W1,
                                                      bf16* __restrict__ W1B) {
    const int li = blockIdx.x * 256 + threadIdx.x;    // over K1*EMB
    const int e  = li & 127;
    const int k  = li >> 7;
    float v = 0.0f;
    if (k < FPAD) {
        if (k < FEAT) v = W1[(size_t)e * (2 * FEAT) + k];
    } else {
        const int kk = k - FPAD;
        if (kk < FEAT) v = W1[(size_t)e * (2 * FEAT) + FEAT + kk];
    }
    W1B[(size_t)k * EMB + e] = (bf16)v;
}

__global__ __launch_bounds__(256) void pack_w2_kernel(const float* __restrict__ W2,
                                                      bf16* __restrict__ W2B) {
    const int li = blockIdx.x * 256 + threadIdx.x;    // over 256*EMB
    const int e  = li & 127;
    const int k  = li >> 7;
    W2B[(size_t)k * EMB + e] = (bf16)W2[(size_t)e * 256 + k];
}

// X2[i][0:128] = H1B[nodes[i]][:], and zero the split-K accumulator buffer.
__global__ __launch_bounds__(256) void pack_x2_kernel(const int* __restrict__ nodes,
                                                      const bf16* __restrict__ H1B,
                                                      bf16* __restrict__ X2,
                                                      float* __restrict__ N2F) {
    const int li = blockIdx.x * 256 + threadIdx.x;    // over BATCH*EMB
    const int i  = li >> 7;
    const int c  = li & 127;
    X2[(size_t)i * 256 + c] = H1B[(size_t)nodes[i] * EMB + c];
    N2F[li] = 0.0f;
}

// X2[i][128:256] = bf16(N2F[i][:] / deg[nodes[i]])
__global__ __launch_bounds__(256) void x2fin_kernel(const int* __restrict__ nodes,
                                                    const float* __restrict__ N2F,
                                                    const float* __restrict__ deg,
                                                    bf16* __restrict__ X2) {
    const int li = blockIdx.x * 256 + threadIdx.x;    // over BATCH*EMB
    const int i  = li >> 7;
    const int c  = li & 127;
    X2[(size_t)i * 256 + EMB + c] = (bf16)(N2F[li] / deg[nodes[i]]);
}

// ---------------------------------------------------------------------------
// GEMM1: neigh1 = (adj @ X1[:, :512]) / deg -> bf16 into X1[:, 512:1024]
// 128x256 block (halves adj HBM traffic vs 128x128; B panel lives in L2),
// double-buffered LDS, fp32->bf16 A conversion on load.
// grid: 128 m-blocks x 2 n-blocks
// ---------------------------------------------------------------------------
__global__ __launch_bounds__(256) void gemm1_kernel(const float* __restrict__ adj,
                                                    bf16* X1,
                                                    const float* __restrict__ deg) {
    __shared__ bf16 As[2][128 * LSTR];
    __shared__ bf16 Bs[2][256 * LSTR];
    const int tid  = threadIdx.x;
    const int wave = tid >> 5;
    const int lane = tid & 31;
    const int half = lane >> 4;
    const int lm   = lane & 15;
    const int mb = (int)(blockIdx.x >> 1) * 128;
    const int nb = (int)(blockIdx.x & 1) * 256;

    auto stage_a = [&](bf16* dst, int kb) {
#pragma unroll
        for (int it = 0; it < 4; ++it) {
            const int li = (it * 256 + tid) * 4;
            const int r = li >> 5, c = li & 31;
            const v4f q = *(const v4f*)(adj + (size_t)(mb + r) * NN + (kb + c));
            v4bf h;
            h[0] = (bf16)q[0]; h[1] = (bf16)q[1];
            h[2] = (bf16)q[2]; h[3] = (bf16)q[3];
            *(v4bf*)(dst + r * LSTR + c) = h;
        }
    };

    v8f acc[16] = {};
    stage_a(As[0], 0);
    stage_b_t<256>(X1, K1, Bs[0], 0, nb, tid);
    int cur = 0;
    for (int kb = 0; kb < NN; kb += 32) {
        __syncthreads();
        if (kb + 32 < NN) {
            stage_a(As[cur ^ 1], kb + 32);
            stage_b_t<256>(X1, K1, Bs[cur ^ 1], kb + 32, nb, tid);
            if (kb + 64 < NN)
                __builtin_prefetch(&adj[(size_t)(mb + (tid & 127)) * NN + (kb + 64)], 0, 1);
        }
        const v16bf a = load_a_frag(As[cur], wave, lane);
        mma8(a, Bs[cur], acc, lane);
        mma8(a, Bs[cur] + 128 * LSTR, acc + 8, lane);
        cur ^= 1;
    }

    float dinv[8];
#pragma unroll
    for (int r = 0; r < 8; ++r)
        dinv[r] = 1.0f / deg[mb + wave * 16 + half * 8 + r];
#pragma unroll
    for (int t = 0; t < 16; ++t) {
        const int col = t * 16 + lm;
#pragma unroll
        for (int r = 0; r < 8; ++r) {
            const int row = wave * 16 + half * 8 + r;
            X1[(size_t)(mb + row) * K1 + FPAD + nb + col] = (bf16)(acc[t][r] * dinv[r]);
        }
    }
}

// ---------------------------------------------------------------------------
// GEMM2: H1 = l2norm(relu(X1 @ W1B + b1)) -> bf16 H1B [NN][128]
// A tile via GLOBAL_LOAD_ASYNC_TO_LDS_B128 (ASYNCcnt-tracked).
// ---------------------------------------------------------------------------
__global__ __launch_bounds__(256) void gemm2_kernel(const bf16* __restrict__ X1,
                                                    const bf16* __restrict__ W1B,
                                                    const float* __restrict__ b1,
                                                    bf16* __restrict__ H1B) {
    __shared__ __align__(16) char smem[65536];
    bf16* As = (bf16*)smem;
    bf16* Bs = As + 128 * LSTR;
    const int tid  = threadIdx.x;
    const int wave = tid >> 5;
    const int lane = tid & 31;
    const int half = lane >> 4;
    const int lm   = lane & 15;
    const int mb = (int)blockIdx.x * 128;

    v8f acc[8] = {};
    for (int kb = 0; kb < K1; kb += 32) {
        __syncthreads();
#if HAVE_ASYNC_LDS
#pragma unroll
        for (int it = 0; it < 2; ++it) {
            const int li = (it * 256 + tid) * 8;
            const int r = li >> 5, c = li & 31;
            __builtin_amdgcn_global_load_async_to_lds_b128(
                (AS1 v4i*)(X1 + (size_t)(mb + r) * K1 + (kb + c)),
                (AS3 v4i*)(As + r * LSTR + c), 0, 0);
        }
#else
        stage_a_bf(X1, K1, As, mb, kb, tid);
#endif
        stage_b_t<128>(W1B, EMB, Bs, kb, 0, tid);
#if HAVE_ASYNC_LDS
        __builtin_amdgcn_s_wait_asynccnt(0);
#endif
        __syncthreads();
        const v16bf a = load_a_frag(As, wave, lane);
        mma8(a, Bs, acc, lane);
    }

    __syncthreads();                    // tiles dead -> reuse LDS as C buffer
    float* Cs = (float*)smem;           // [128][128]
#pragma unroll
    for (int t = 0; t < 8; ++t) {
        const int col = t * 16 + lm;
#pragma unroll
        for (int r = 0; r < 8; ++r) {
            const int row = wave * 16 + half * 8 + r;
            Cs[row * 128 + col] = fmaxf(acc[t][r] + b1[col], 0.0f);
        }
    }
    __syncthreads();
    if (tid < 128) {
        float s = 0.0f;
        for (int c = 0; c < 128; ++c) {
            const float x = Cs[tid * 128 + c];
            s += x * x;
        }
        const float inv = 1.0f / fmaxf(sqrtf(s), 1e-12f);
        for (int c = 0; c < 128; ++c)
            H1B[(size_t)(mb + tid) * EMB + c] = (bf16)(Cs[tid * 128 + c] * inv);
    }
}

// ---------------------------------------------------------------------------
// GEMM3: N2F += adj[nodes] @ H1B   (split-K: 8 chunks of 2048 for occupancy;
// fp32 atomic accumulation). grid: 32 m-blocks x 8 k-chunks = 256 blocks
// ---------------------------------------------------------------------------
__global__ __launch_bounds__(256) void gemm3_kernel(const int* __restrict__ nodes,
                                                    const float* __restrict__ adj,
                                                    const bf16* __restrict__ H1B,
                                                    float* __restrict__ N2F) {
    __shared__ bf16 As[2][128 * LSTR];
    __shared__ bf16 Bs[2][128 * LSTR];
    const int tid  = threadIdx.x;
    const int wave = tid >> 5;
    const int lane = tid & 31;
    const int half = lane >> 4;
    const int lm   = lane & 15;
    const int mb = (int)(blockIdx.x >> 3) * 128;
    const int kc = (int)(blockIdx.x & 7) * KCH;

    auto stage_a = [&](bf16* dst, int kb) {
#pragma unroll
        for (int it = 0; it < 4; ++it) {
            const int li = (it * 256 + tid) * 4;
            const int r = li >> 5, c = li & 31;
            const int nd = nodes[mb + r];
            const v4f q = *(const v4f*)(adj + (size_t)nd * NN + (kb + c));
            v4bf h;
            h[0] = (bf16)q[0]; h[1] = (bf16)q[1];
            h[2] = (bf16)q[2]; h[3] = (bf16)q[3];
            *(v4bf*)(dst + r * LSTR + c) = h;
        }
    };

    v8f acc[8] = {};
    stage_a(As[0], kc);
    stage_b_t<128>(H1B, EMB, Bs[0], kc, 0, tid);
    int cur = 0;
    for (int kb = kc; kb < kc + KCH; kb += 32) {
        __syncthreads();
        if (kb + 32 < kc + KCH) {
            stage_a(As[cur ^ 1], kb + 32);
            stage_b_t<128>(H1B, EMB, Bs[cur ^ 1], kb + 32, 0, tid);
            __builtin_prefetch(&adj[(size_t)nodes[mb + (tid & 127)] * NN + (kb + 64)], 0, 1);
        }
        const v16bf a = load_a_frag(As[cur], wave, lane);
        mma8(a, Bs[cur], acc, lane);
        cur ^= 1;
    }

#pragma unroll
    for (int t = 0; t < 8; ++t) {
        const int col = t * 16 + lm;
#pragma unroll
        for (int r = 0; r < 8; ++r) {
            const int row = wave * 16 + half * 8 + r;
            atomicAdd(&N2F[(size_t)(mb + row) * EMB + col], acc[t][r]);
        }
    }
}

// ---------------------------------------------------------------------------
// GEMM4: H2 = l2norm(relu(X2 @ W2B + b2)) -> fp32 H2 [BATCH][128]
// grid: 32 m-blocks, K=256; async A tile
// ---------------------------------------------------------------------------
__global__ __launch_bounds__(256) void gemm4_kernel(const bf16* __restrict__ X2,
                                                    const bf16* __restrict__ W2B,
                                                    const float* __restrict__ b2,
                                                    float* __restrict__ H2) {
    __shared__ __align__(16) char smem[65536];
    bf16* As = (bf16*)smem;
    bf16* Bs = As + 128 * LSTR;
    const int tid  = threadIdx.x;
    const int wave = tid >> 5;
    const int lane = tid & 31;
    const int half = lane >> 4;
    const int lm   = lane & 15;
    const int mb = (int)blockIdx.x * 128;

    v8f acc[8] = {};
    for (int kb = 0; kb < 256; kb += 32) {
        __syncthreads();
#if HAVE_ASYNC_LDS
#pragma unroll
        for (int it = 0; it < 2; ++it) {
            const int li = (it * 256 + tid) * 8;
            const int r = li >> 5, c = li & 31;
            __builtin_amdgcn_global_load_async_to_lds_b128(
                (AS1 v4i*)(X2 + (size_t)(mb + r) * 256 + (kb + c)),
                (AS3 v4i*)(As + r * LSTR + c), 0, 0);
        }
#else
        stage_a_bf(X2, 256, As, mb, kb, tid);
#endif
        stage_b_t<128>(W2B, EMB, Bs, kb, 0, tid);
#if HAVE_ASYNC_LDS
        __builtin_amdgcn_s_wait_asynccnt(0);
#endif
        __syncthreads();
        const v16bf a = load_a_frag(As, wave, lane);
        mma8(a, Bs, acc, lane);
    }

    __syncthreads();
    float* Cs = (float*)smem;
#pragma unroll
    for (int t = 0; t < 8; ++t) {
        const int col = t * 16 + lm;
#pragma unroll
        for (int r = 0; r < 8; ++r) {
            const int row = wave * 16 + half * 8 + r;
            Cs[row * 128 + col] = fmaxf(acc[t][r] + b2[col], 0.0f);
        }
    }
    __syncthreads();
    if (tid < 128) {
        float s = 0.0f;
        for (int c = 0; c < 128; ++c) {
            const float x = Cs[tid * 128 + c];
            s += x * x;
        }
        const float inv = 1.0f / fmaxf(sqrtf(s), 1e-12f);
        for (int c = 0; c < 128; ++c)
            H2[(size_t)(mb + tid) * EMB + c] = Cs[tid * 128 + c] * inv;
    }
}

// X3 = H2 @ Wl1^T + bl1  (fp32 VALU, 134 MFLOP)
__global__ __launch_bounds__(256) void lin1_kernel(const float* __restrict__ H2,
                                                   const float* __restrict__ Wl1,
                                                   const float* __restrict__ bl1,
                                                   float* __restrict__ X3) {
    const int li = blockIdx.x * 256 + threadIdx.x;    // over BATCH*128
    const int i  = li >> 7;
    const int j  = li & 127;
    float s = bl1[j];
#pragma unroll 4
    for (int k = 0; k < 128; ++k)
        s += H2[(size_t)i * 128 + k] * Wl1[(size_t)j * 128 + k];
    X3[li] = s;
}

// out = softmax(X3 @ Wl2^T + bl2)
__global__ __launch_bounds__(256) void lin2_softmax_kernel(const float* __restrict__ X3,
                                                           const float* __restrict__ Wl2,
                                                           const float* __restrict__ bl2,
                                                           float* __restrict__ out) {
    const int i = blockIdx.x * 256 + threadIdx.x;     // over BATCH
    float lg[NCLS];
    float mx = -1e30f;
#pragma unroll
    for (int c = 0; c < NCLS; ++c) {
        float s = bl2[c];
        for (int k = 0; k < 128; ++k)
            s += X3[(size_t)i * 128 + k] * Wl2[(size_t)c * 128 + k];
        lg[c] = s;
        mx = fmaxf(mx, s);
    }
    float sum = 0.0f;
#pragma unroll
    for (int c = 0; c < NCLS; ++c) {
        lg[c] = __expf(lg[c] - mx);
        sum += lg[c];
    }
    const float inv = 1.0f / sum;
#pragma unroll
    for (int c = 0; c < NCLS; ++c)
        out[(size_t)i * NCLS + c] = lg[c] * inv;
}

// ---------------------------------------------------------------------------
extern "C" void kernel_launch(void* const* d_in, const int* in_sizes, int n_in,
                              void* d_out, int out_size, void* d_ws, size_t ws_size,
                              hipStream_t stream) {
    (void)in_sizes; (void)n_in; (void)out_size; (void)ws_size;
    const int*   nodes = (const int*)  d_in[0];
    const float* adj   = (const float*)d_in[1];
    const float* data  = (const float*)d_in[2];
    const float* W1    = (const float*)d_in[3];
    const float* b1    = (const float*)d_in[4];
    const float* W2    = (const float*)d_in[5];
    const float* b2    = (const float*)d_in[6];
    const float* Wl1   = (const float*)d_in[7];
    const float* bl1   = (const float*)d_in[8];
    const float* Wl2   = (const float*)d_in[9];
    const float* bl2   = (const float*)d_in[10];
    float* out = (float*)d_out;

    char* ws = (char*)d_ws;
    size_t off = 0;
    float* deg = (float*)(ws + off); off += (size_t)NN * 4;
    bf16*  X1  = (bf16*) (ws + off); off += (size_t)NN * K1 * 2;      // [data|neigh1]
    bf16*  W1B = (bf16*) (ws + off); off += (size_t)K1 * EMB * 2;
    bf16*  H1B = (bf16*) (ws + off); off += (size_t)NN * EMB * 2;
    bf16*  X2  = (bf16*) (ws + off); off += (size_t)BATCH * 256 * 2;  // [H1[nodes]|neigh2]
    bf16*  W2B = (bf16*) (ws + off); off += (size_t)256 * EMB * 2;
    float* H2  = (float*)(ws + off); off += (size_t)BATCH * EMB * 4;
    float* X3  = (float*)(ws + off); off += (size_t)BATCH * EMB * 4;
    float* N2F = (float*)(ws + off); off += (size_t)BATCH * EMB * 4;  // split-K acc

    deg_kernel      <<<NN, 256, 0, stream>>>(adj, deg);
    pack_data_kernel<<<(NN * FPAD) / 256, 256, 0, stream>>>(data, X1);
    pack_w1_kernel  <<<(K1 * EMB) / 256, 256, 0, stream>>>(W1, W1B);
    pack_w2_kernel  <<<(256 * EMB) / 256, 256, 0, stream>>>(W2, W2B);
    gemm1_kernel    <<<256, 256, 0, stream>>>(adj, X1, deg);
    gemm2_kernel    <<<128, 256, 0, stream>>>(X1, W1B, b1, H1B);
    pack_x2_kernel  <<<(BATCH * EMB) / 256, 256, 0, stream>>>(nodes, H1B, X2, N2F);
    gemm3_kernel    <<<256, 256, 0, stream>>>(nodes, adj, H1B, N2F);
    x2fin_kernel    <<<(BATCH * EMB) / 256, 256, 0, stream>>>(nodes, N2F, deg, X2);
    gemm4_kernel    <<<32, 256, 0, stream>>>(X2, W2B, b2, H2);
    lin1_kernel     <<<(BATCH * EMB) / 256, 256, 0, stream>>>(H2, Wl1, bl1, X3);
    lin2_softmax_kernel<<<BATCH / 256, 256, 0, stream>>>(X3, Wl2, bl2, out);
}
